// MoELayer_28827820491317
// MI455X (gfx1250) — matrix operand order repaired and checked by reading, not compiled, not run on hardware
//
#include <hip/hip_runtime.h>
#include <hip/hip_bf16.h>

typedef __bf16 bf16_t;
typedef __attribute__((ext_vector_type(16))) __bf16 v16bf;
typedef __attribute__((ext_vector_type(8)))  __bf16 v8bf;
typedef __attribute__((ext_vector_type(4)))  __bf16 v4bf;
typedef __attribute__((ext_vector_type(8)))  float  v8f;
typedef __attribute__((ext_vector_type(4)))  float  v4f;

#define T_TOKENS 128
#define DDIM     512
#define HDIM     1024
#define NEXP     8
#define BSTRIDE  40   // LDS row stride in bf16 (80B: 16B-aligned rows, conflict-reducing pad)

// Branch-free tanh-approx gelu (jax.nn.gelu approximate=True).
// tanh(u) = 1 - 2/(exp(2u)+1); exp over/underflow saturates correctly -> no branches.
__device__ __forceinline__ float gelu_tanh(float y) {
    float u = 0.7978845608028654f * (y + 0.044715f * y * y * y);
    float t = __expf(2.0f * u);
    float th = 1.0f - 2.0f * __builtin_amdgcn_rcpf(t + 1.0f);
    return 0.5f * y * (1.0f + th);
}

// ---- WMMA operand loaders (wave32, v_wmma_f32_16x16x32_bf16) ----
// A (16x32 bf16): lane L = row M=L&15; element j: K = (j>>3)*16 + (L>>4)*8 + (j&7)
__device__ __forceinline__ v16bf load_a_tile(const bf16_t* __restrict__ src,
                                             int ld, int r0, int k0, int lane) {
    const bf16_t* p = src + (size_t)(r0 + (lane & 15)) * ld + k0 + ((lane >> 4) << 3);
    v8bf lo = *(const v8bf*)(p);
    v8bf hi = *(const v8bf*)(p + 16);
    return __builtin_shufflevector(lo, hi, 0,1,2,3,4,5,6,7,8,9,10,11,12,13,14,15);
}

// B (32x16 bf16) from LDS stage: lane L = col N=L&15; element j: K = (j>>3)*16 + (L>>4)*8 + (j&7)
__device__ __forceinline__ v16bf load_b_lds(const bf16_t* p) {
    v16bf r;
#pragma unroll
    for (int j = 0; j < 8; ++j) {
        r[j] = p[j * BSTRIDE];
        r[8 + j] = p[(16 + j) * BSTRIDE];
    }
    return r;
}

// Cooperative fp32 -> bf16 stage of a 32(K) x 32(N) weight tile into LDS.
// 256 threads: thread -> (row = tid>>3, 4 cols at (tid&7)*4); one 16B global load each.
__device__ __forceinline__ void stage_b_tile(bf16_t* lds,  // [32][BSTRIDE]
                                             const float* __restrict__ src,
                                             int ld, int k0, int n0, int tid) {
    const int row = tid >> 3;
    const int c4  = (tid & 7) << 2;
    v4f f = *(const v4f*)(src + (size_t)(k0 + row) * ld + n0 + c4);
    v4bf b;
    b[0] = (bf16_t)f[0]; b[1] = (bf16_t)f[1];
    b[2] = (bf16_t)f[2]; b[3] = (bf16_t)f[3];
    *(v4bf*)(lds + row * BSTRIDE + c4) = b;
}

// ---------------- Kernel 1: token convert + router (softmax + top-2) ----------------
__global__ void moe_router_kernel(const float* __restrict__ inputs,   // [T,D]
                                  const float* __restrict__ rw,       // [D,E]
                                  bf16_t* __restrict__ tokb,          // [T,D] bf16
                                  float* __restrict__ gate_full) {    // [T,E] dense gate
    const int tid = threadIdx.x;
    for (int i = tid; i < T_TOKENS * DDIM; i += 256) tokb[i] = (bf16_t)inputs[i];

    if (tid < T_TOKENS) {
        float l[NEXP];
#pragma unroll
        for (int e = 0; e < NEXP; ++e) l[e] = 0.0f;
        const float* x = inputs + (size_t)tid * DDIM;
        for (int d = 0; d < DDIM; ++d) {
            float xv = x[d];
            const float* r = rw + (size_t)d * NEXP;
#pragma unroll
            for (int e = 0; e < NEXP; ++e) l[e] += xv * r[e];
        }
        float m = l[0];
#pragma unroll
        for (int e = 1; e < NEXP; ++e) m = fmaxf(m, l[e]);
        float s = 0.0f;
        float p[NEXP];
#pragma unroll
        for (int e = 0; e < NEXP; ++e) { p[e] = expf(l[e] - m); s += p[e]; }
        float inv = 1.0f / s;
#pragma unroll
        for (int e = 0; e < NEXP; ++e) p[e] *= inv;
        int i0 = 0;
#pragma unroll
        for (int e = 1; e < NEXP; ++e) if (p[e] > p[i0]) i0 = e;
        int i1 = (i0 == 0) ? 1 : 0;
#pragma unroll
        for (int e = 0; e < NEXP; ++e) if (e != i0 && p[e] > p[i1]) i1 = e;
#pragma unroll
        for (int e = 0; e < NEXP; ++e)
            gate_full[tid * NEXP + e] = (e == i0 || e == i1) ? p[e] : 0.0f;
    }
}

// ---------------- Kernel 2: act[e,t,h] = (tok@Wv*sv) * gelu(tok@W*sg) ----------------
// grid: (HDIM/32, NEXP); block 256 = 8 waves; wave w -> token rows [16w,16w+16).
// 32-wide h-tile; B tiles double-buffered in LDS (one barrier per K-chunk).
__global__ void moe_act_kernel(const bf16_t* __restrict__ tokb,  // [T,D] bf16
                               const float*  __restrict__ wv,    // [E,D,H]
                               const float*  __restrict__ wvs,   // [E,1,H]
                               const float*  __restrict__ wg,    // [E,D,H]
                               const float*  __restrict__ wgs,   // [E,1,H]
                               bf16_t* __restrict__ act) {       // [E,T,H] bf16
    __shared__ __align__(16) bf16_t lbv[2][32 * BSTRIDE];
    __shared__ __align__(16) bf16_t lbg[2][32 * BSTRIDE];

    const int tid  = threadIdx.x;
    const int lane = tid & 31;
    const int wave = tid >> 5;
    const int h0 = blockIdx.x * 32;
    const int e  = blockIdx.y;
    const int t0 = wave * 16;

    const float* wv_e = wv + (size_t)e * DDIM * HDIM;
    const float* wg_e = wg + (size_t)e * DDIM * HDIM;

    v8f accV[2] = {};
    v8f accG[2] = {};

    const int NC = DDIM / 32;  // 16 K-chunks
    stage_b_tile(lbv[0], wv_e, HDIM, 0, h0, tid);
    stage_b_tile(lbg[0], wg_e, HDIM, 0, h0, tid);

    for (int i = 0; i < NC; ++i) {
        __syncthreads();  // chunk i staged; prior reads of buf (i+1)&1 are done
        const int cur = i & 1;
        if (i + 1 < NC) {
            stage_b_tile(lbv[cur ^ 1], wv_e, HDIM, (i + 1) * 32, h0, tid);
            stage_b_tile(lbg[cur ^ 1], wg_e, HDIM, (i + 1) * 32, h0, tid);
        }
        if (i + 2 < NC) {  // prefetch chunk i+2 of the weight stream
            __builtin_prefetch(wv_e + (size_t)((i + 2) * 32 + (tid >> 3)) * HDIM + h0, 0, 2);
            __builtin_prefetch(wg_e + (size_t)((i + 2) * 32 + (tid >> 3)) * HDIM + h0, 0, 2);
        }
        v16bf a = load_a_tile(tokb, DDIM, t0, i * 32, lane);
#pragma unroll
        for (int ns = 0; ns < 2; ++ns) {
            v16bf bv = load_b_lds(&lbv[cur][((lane >> 4) << 3) * BSTRIDE + ns * 16 + (lane & 15)]);
            v16bf bg = load_b_lds(&lbg[cur][((lane >> 4) << 3) * BSTRIDE + ns * 16 + (lane & 15)]);
            accV[ns] = __builtin_amdgcn_wmma_f32_16x16x32_bf16(false, a, false, bv,
                                                               (short)0, accV[ns], false, false);
            accG[ns] = __builtin_amdgcn_wmma_f32_16x16x32_bf16(false, a, false, bg,
                                                               (short)0, accG[ns], false, false);
        }
    }

    // C layout: lane -> column N=lane&15; VGPR v -> row M = v + 8*(lane>>4)
    const int n = lane & 15;
    const int mbase = (lane >> 4) << 3;
#pragma unroll
    for (int ns = 0; ns < 2; ++ns) {
        const int h = h0 + ns * 16 + n;
        const float sv = wvs[e * HDIM + h];   // column dequant scales folded post-GEMM
        const float sg = wgs[e * HDIM + h];
#pragma unroll
        for (int v = 0; v < 8; ++v) {
            int t = t0 + mbase + v;
            float xv = accV[ns][v] * sv;
            float yv = accG[ns][v] * sg;
            act[((size_t)e * T_TOKENS + t) * HDIM + h] = (bf16_t)(xv * gelu_tanh(yv));
        }
    }
}

// ---------------- Kernel 3: out[t,d] = sum_e gate[t,e]*s1[e,d]*(act_e @ W1_e)[t,d] ------
// grid: DDIM/32; block 256 = 8 waves; wave w -> token rows [16w,16w+16).
// Flattened expert*K pipeline (256 chunks), W1 double-buffered in LDS.
__global__ void moe_out_kernel(const bf16_t* __restrict__ act,   // [E,T,H] bf16
                               const float*  __restrict__ w1,    // [E,H,D]
                               const float*  __restrict__ w1s,   // [E,1,D]
                               const float*  __restrict__ gate,  // [T,E]
                               bf16_t* __restrict__ out) {       // [T,D] bf16
    __shared__ __align__(16) bf16_t lb1[2][32 * BSTRIDE];

    const int tid  = threadIdx.x;
    const int lane = tid & 31;
    const int wave = tid >> 5;
    const int d0 = blockIdx.x * 32;
    const int t0 = wave * 16;
    const int n = lane & 15;
    const int mbase = (lane >> 4) << 3;

    const int KCH = HDIM / 32;        // 32 K-chunks per expert
    const int NC  = NEXP * KCH;       // 256 chunks total

    float ctot[2][8] = {};
    v8f acc[2] = {};
    const v8f vzero = {};

    // chunk c -> expert e = c/KCH, k0 = (c%KCH)*32
    stage_b_tile(lb1[0], w1, DDIM, 0, d0, tid);  // chunk 0 (expert 0, k0=0)

    for (int c = 0; c < NC; ++c) {
        __syncthreads();  // chunk c staged; prior reads of buf (c+1)&1 are done
        const int cur = c & 1;
        if (c + 1 < NC) {
            const int e1 = (c + 1) >> 5, k1 = ((c + 1) & 31) << 5;
            stage_b_tile(lb1[cur ^ 1], w1 + (size_t)e1 * HDIM * DDIM, DDIM, k1, d0, tid);
        }
        if (c + 2 < NC) {
            const int e2 = (c + 2) >> 5, k2 = ((c + 2) & 31) << 5;
            __builtin_prefetch(w1 + (size_t)e2 * HDIM * DDIM +
                               (size_t)(k2 + (tid >> 3)) * DDIM + d0, 0, 2);
        }
        const int e  = c >> 5;
        const int k0 = (c & 31) << 5;
        const bf16_t* act_e = act + (size_t)e * T_TOKENS * HDIM;
        v16bf a = load_a_tile(act_e, HDIM, t0, k0, lane);
#pragma unroll
        for (int ns = 0; ns < 2; ++ns) {
            v16bf b = load_b_lds(&lb1[cur][((lane >> 4) << 3) * BSTRIDE + ns * 16 + (lane & 15)]);
            acc[ns] = __builtin_amdgcn_wmma_f32_16x16x32_bf16(false, a, false, b,
                                                              (short)0, acc[ns], false, false);
        }
        if ((c & 31) == 31) {  // expert finished: fold gate * w1_scale, reset accumulators
#pragma unroll
            for (int ns = 0; ns < 2; ++ns) {
                const float s1 = w1s[e * DDIM + d0 + ns * 16 + n];
#pragma unroll
                for (int v = 0; v < 8; ++v) {
                    int t = t0 + mbase + v;
                    ctot[ns][v] += gate[t * NEXP + e] * s1 * acc[ns][v];
                }
            }
            acc[0] = vzero;
            acc[1] = vzero;
        }
    }
#pragma unroll
    for (int ns = 0; ns < 2; ++ns)
#pragma unroll
        for (int v = 0; v < 8; ++v) {
            int t = t0 + mbase + v;
            out[(size_t)t * DDIM + d0 + ns * 16 + n] = (bf16_t)ctot[ns][v];
        }
}

extern "C" void kernel_launch(void* const* d_in, const int* in_sizes, int n_in,
                              void* d_out, int out_size, void* d_ws, size_t ws_size,
                              hipStream_t stream) {
    const float* inputs   = (const float*)d_in[0];   // [1,128,512]
    const float* router_w = (const float*)d_in[1];   // [512,8]
    const float* wv_w     = (const float*)d_in[2];   // [8,512,1024]
    const float* wv_s     = (const float*)d_in[3];   // [8,1,1024]
    const float* w_w      = (const float*)d_in[4];   // [8,512,1024]
    const float* w_s      = (const float*)d_in[5];   // [8,1,1024]
    const float* w1_w     = (const float*)d_in[6];   // [8,1024,512]
    const float* w1_s     = (const float*)d_in[7];   // [8,1,512]
    bf16_t* out = (bf16_t*)d_out;                    // [1,128,512] bf16

    char* ws = (char*)d_ws;
    bf16_t* tokb      = (bf16_t*)ws;                          // 128 KB
    float*  gate_full = (float*)(ws + 131072);                // 4 KB
    bf16_t* act       = (bf16_t*)(ws + 131072 + 4096);        // 2 MB

    moe_router_kernel<<<1, 256, 0, stream>>>(inputs, router_w, tokb, gate_full);
    moe_act_kernel<<<dim3(HDIM / 32, NEXP), 256, 0, stream>>>(tokb, wv_w, wv_s, w_w, w_s, act);
    moe_out_kernel<<<dim3(DDIM / 32), 256, 0, stream>>>(act, w1_w, w1_s, gate_full, out);
}
